// MultiHeadAttention_6562710028653
// MI455X (gfx1250) — matrix operand order repaired and checked by reading, not compiled, not run on hardware
//
#include <hip/hip_runtime.h>
#include <cstdint>
#include <cstddef>

// ---------------------------------------------------------------------------
// MHA forward for MI455X (gfx1250): bf16 WMMA GEMMs + fused flash attention.
// d_out = [ out (B*S*D fp32) | qk (B*H*S*S fp32, -inf masked logits) ]
// d_ws  : 4 x B*S*D bf16 buffers (Q,K,V,attn) = 64 MB.
// ---------------------------------------------------------------------------

#if defined(__has_builtin)
#if __has_builtin(__builtin_amdgcn_global_load_async_to_lds_b128) && \
    __has_builtin(__builtin_amdgcn_s_wait_asynccnt)
#define USE_ASYNC_LDS 1
#endif
#endif
#ifndef USE_ASYNC_LDS
#define USE_ASYNC_LDS 0
#endif

typedef __attribute__((ext_vector_type(16))) __bf16 v16bf;
typedef __attribute__((ext_vector_type(8)))  float  v8f;

#define Bx  8
#define Sx  1024
#define Dx  1024
#define Hx  16
#define DHx 64

__device__ __forceinline__ unsigned short f2bf(float f) {
  unsigned int u = __float_as_uint(f);
  unsigned int r = u + 0x7FFFu + ((u >> 16) & 1u);   // round-to-nearest-even
  return (unsigned short)(r >> 16);
}
__device__ __forceinline__ unsigned int pack2bf(float a, float b) {
  return (unsigned int)f2bf(a) | ((unsigned int)f2bf(b) << 16);
}

union AFrag { v16bf v; uint4 u[2]; };

__device__ __forceinline__ v8f wmma_bf16(const AFrag& a, const AFrag& b, v8f c) {
  return __builtin_amdgcn_wmma_f32_16x16x32_bf16(false, a.v, false, b.v,
                                                 (short)0, c, false, false);
}

#if USE_ASYNC_LDS
// Async global->LDS b128 copy (CDNA5 ASYNCcnt path); 16 bytes per lane.
// Builtin signature (per hipcc diagnostic): (v4i as1* src, v4i as3* dst,
// imm offset, imm cpol); offset applies to both addresses (ISA 08 §4.4).
typedef int v4i_vs __attribute__((vector_size(16)));
template<int OFF>
__device__ __forceinline__ void async_copy_b128(const void* gsrc, void* ldst) {
  __builtin_amdgcn_global_load_async_to_lds_b128(
      (__attribute__((address_space(1))) v4i_vs*)gsrc,
      (__attribute__((address_space(3))) v4i_vs*)ldst, OFF, 0);
}
#endif

// ---------------------------------------------------------------------------
// GEMM: Y[M,N] = X[M,K] @ W[N,K]^T (+bias) * scale.  M=8192, N=K=1024.
// Block tile 128(M) x 64(N), 256 threads = 8 waves, K-step 32 (bf16 WMMA).
// ---------------------------------------------------------------------------
template<bool IN_BF16, bool OUT_F32>
__global__ void __launch_bounds__(256)
gemm_bias_kernel(const void* __restrict__ Xv, const float* __restrict__ W,
                 const float* __restrict__ bias, void* __restrict__ Yv,
                 float scale)
{
  constexpr int Kd = 1024, Nd = 1024;
  __shared__ unsigned short Asg[128 * 40];   // 128 x 32 bf16, pitch 40 (bank-safe)
  __shared__ unsigned short Bsg[64 * 40];    // 64  x 32 bf16

  const int tid  = threadIdx.x;
  const int w    = tid >> 5;
  const int lane = tid & 31;
  const int half = lane >> 4;
  const int l16  = lane & 15;
  const int n0   = blockIdx.x * 64;
  const int m0   = blockIdx.y * 128;

  v8f acc[4];
#pragma unroll
  for (int jn = 0; jn < 4; ++jn) acc[jn] = (v8f){0,0,0,0,0,0,0,0};

  for (int k0 = 0; k0 < Kd; k0 += 32) {
    __syncthreads();
    // --- stage A tile (128x32) -> bf16 LDS ---
    {
      const int row = tid >> 1, cb = (tid & 1) * 16;
      if constexpr (IN_BF16) {
        const unsigned short* srcp = (const unsigned short*)Xv +
                                     (size_t)(m0 + row) * Kd + k0 + cb;
#if USE_ASYNC_LDS
        async_copy_b128<0>(srcp, &Asg[row * 40 + cb]);
        async_copy_b128<16>(srcp, &Asg[row * 40 + cb]);  // offset on both sides
#else
        const uint4* src = (const uint4*)srcp;
        *(uint4*)&Asg[row * 40 + cb]     = src[0];
        *(uint4*)&Asg[row * 40 + cb + 8] = src[1];
#endif
        if (k0 + 32 < Kd) __builtin_prefetch(srcp + 32, 0, 3);
      } else {
        const float4* src = (const float4*)((const float*)Xv +
                                            (size_t)(m0 + row) * Kd + k0 + cb);
        unsigned int* dst = (unsigned int*)&Asg[row * 40 + cb];
#pragma unroll
        for (int e = 0; e < 4; ++e) {
          float4 f = src[e];
          dst[2 * e]     = pack2bf(f.x, f.y);
          dst[2 * e + 1] = pack2bf(f.z, f.w);
        }
        if (k0 + 32 < Kd) __builtin_prefetch(src + 8, 0, 3);   // next K-step
      }
    }
    // --- stage B tile (64 rows of W x 32) -> bf16 LDS ---
    {
      const int row = tid >> 2, cb = (tid & 3) * 8;
      const float4* src = (const float4*)(W + (size_t)(n0 + row) * Kd + k0 + cb);
      unsigned int* dst = (unsigned int*)&Bsg[row * 40 + cb];
#pragma unroll
      for (int e = 0; e < 2; ++e) {
        float4 f = src[e];
        dst[2 * e]     = pack2bf(f.x, f.y);
        dst[2 * e + 1] = pack2bf(f.z, f.w);
      }
      if (k0 + 32 < Kd) __builtin_prefetch(src + 4, 0, 3);     // next K-step
    }
#if USE_ASYNC_LDS
    if constexpr (IN_BF16) __builtin_amdgcn_s_wait_asynccnt(0);
#endif
    __syncthreads();

    // --- A fragment: wave w owns rows [w*16, w*16+16) ---
    AFrag a;
    {
      const unsigned short* ap = &Asg[(w * 16 + l16) * 40 + half * 8];
      a.u[0] = *(const uint4*)ap;          // K 0..7  / 8..15
      a.u[1] = *(const uint4*)(ap + 16);   // K 16..23 / 24..31
    }
#pragma unroll
    for (int jn = 0; jn < 4; ++jn) {
      AFrag bf;
      const unsigned short* bp = &Bsg[(jn * 16 + l16) * 40 + half * 8];
      bf.u[0] = *(const uint4*)bp;
      bf.u[1] = *(const uint4*)(bp + 16);
      acc[jn] = wmma_bf16(a, bf, acc[jn]);
    }
  }

  // --- epilogue: bias + scale, store ---
#pragma unroll
  for (int jn = 0; jn < 4; ++jn) {
    const int col = n0 + jn * 16 + l16;
    const float bn = bias ? bias[col] : 0.0f;
#pragma unroll
    for (int v = 0; v < 8; ++v) {
      const int row = m0 + w * 16 + v + half * 8;
      const float val = (acc[jn][v] + bn) * scale;
      if (OUT_F32)
        ((float*)Yv)[(size_t)row * Nd + col] = val;
      else
        ((unsigned short*)Yv)[(size_t)row * Nd + col] = f2bf(val);
    }
  }
}

// ---------------------------------------------------------------------------
// Fill qk with -inf (columns beyond each block's causal band are never touched
// by the attention kernel).
// ---------------------------------------------------------------------------
__global__ void __launch_bounds__(256)
fill_neginf_kernel(float4* __restrict__ p)
{
  const size_t i = (size_t)blockIdx.x * blockDim.x + threadIdx.x;
  const float ninf = -__builtin_inff();
  const float4 f = {ninf, ninf, ninf, ninf};
  p[i * 2]     = f;
  p[i * 2 + 1] = f;
}

// ---------------------------------------------------------------------------
// Fused flash attention.  One block = (b, h, 128 q rows), 8 waves, each wave
// owns a 16-row q tile.  Q/K pre-scaled by DH^-0.25 in ws.  Writes qk logits
// (fp32 + mask) once, accumulates O with online softmax, stores attn bf16.
// ---------------------------------------------------------------------------
__global__ void __launch_bounds__(256)
flash_attn_kernel(const unsigned short* __restrict__ Qw,
                  const unsigned short* __restrict__ Kw,
                  const unsigned short* __restrict__ Vw,
                  float* __restrict__ qk,
                  unsigned short* __restrict__ Attn)
{
  __shared__ unsigned short Klds[32 * 72];       // [k][d] 32x64 bf16, pitch 72
  __shared__ unsigned short Vlds[64 * 40];       // transposed [d][k] 64x32, pitch 40
  __shared__ unsigned short Plds[8 * 16 * 40];   // per-wave P 16x32, pitch 40

  const int tid  = threadIdx.x;
  const int w    = tid >> 5;
  const int lane = tid & 31;
  const int half = lane >> 4;
  const int l16  = lane & 15;

  const int qbIdx = blockIdx.x & 7;
  const int h     = (blockIdx.x >> 3) & 15;
  const int b     = blockIdx.x >> 7;
  const int qrow0 = qbIdx * 128 + w * 16;

  const float NINF = -__builtin_inff();

  // --- Q fragments (16x64 -> two 16x32 A-frags), loaded once ---
  AFrag qa[2];
  {
    const uint4* q4 = (const uint4*)(Qw + ((size_t)(b * Sx + qrow0 + l16) * Dx + h * DHx));
#pragma unroll
    for (int j = 0; j < 2; ++j) {
      qa[j].u[0] = q4[j * 4 + half];
      qa[j].u[1] = q4[j * 4 + half + 2];
    }
  }

  float mrun[8], lrun[8];
#pragma unroll
  for (int v = 0; v < 8; ++v) { mrun[v] = NINF; lrun[v] = 0.0f; }
  v8f o[4];
#pragma unroll
  for (int jn = 0; jn < 4; ++jn) o[jn] = (v8f){0,0,0,0,0,0,0,0};

  const int nIter = (qbIdx + 1) * 4;             // 32 k-rows per iteration
  const size_t qkBase = (size_t)(b * Hx + h) * Sx * Sx;

  for (int it = 0; it < nIter; ++it) {
    const int k0 = it * 32;
    __syncthreads();
    // --- cooperatively stage K (async, row-major) and V (transposed) ---
    {
      const int r = tid >> 3, c = (tid & 7) * 8;
      const size_t base = (size_t)(b * Sx + k0 + r) * Dx + h * DHx + c;
#if USE_ASYNC_LDS
      async_copy_b128<0>(Kw + base, &Klds[r * 72 + c]);
#else
      uint4 kv = *(const uint4*)(Kw + base);
      *(uint4*)&Klds[r * 72 + c] = kv;
#endif
      uint4 vv = *(const uint4*)(Vw + base);
      const unsigned short* pv = (const unsigned short*)&vv;
#pragma unroll
      for (int e = 0; e < 8; ++e) Vlds[(c + e) * 40 + r] = pv[e];

      if (it + 1 < nIter) {                       // prefetch next k-tile rows
        __builtin_prefetch(Kw + base + (size_t)32 * Dx, 0, 3);
        __builtin_prefetch(Vw + base + (size_t)32 * Dx, 0, 3);
      }
    }
#if USE_ASYNC_LDS
    __builtin_amdgcn_s_wait_asynccnt(0);
#endif
    __syncthreads();

    // --- scores: two 16x16 tiles, each = 2 chained WMMAs over d ---
    v8f sc[2];
#pragma unroll
    for (int s = 0; s < 2; ++s) {
      v8f z = (v8f){0,0,0,0,0,0,0,0};
#pragma unroll
      for (int dj = 0; dj < 2; ++dj) {
        AFrag kb;
        const unsigned short* kp = &Klds[(s * 16 + l16) * 72 + dj * 32 + half * 8];
        kb.u[0] = *(const uint4*)kp;
        kb.u[1] = *(const uint4*)(kp + 16);
        z = wmma_bf16(qa[dj], kb, z);
      }
      sc[s] = z;
    }

    // --- mask, write qk, online softmax update ---
    float corr[8];
#pragma unroll
    for (int v = 0; v < 8; ++v) {
      const int row = qrow0 + v + half * 8;
      const int c0 = k0 + l16, c1 = k0 + 16 + l16;
      float f0 = (c0 <= row) ? sc[0][v] : NINF;
      float f1 = (c1 <= row) ? sc[1][v] : NINF;
      qk[qkBase + (size_t)row * Sx + c0] = f0;
      qk[qkBase + (size_t)row * Sx + c1] = f1;

      float t = fmaxf(f0, f1);
      t = fmaxf(t, __shfl_xor(t, 1, 32));
      t = fmaxf(t, __shfl_xor(t, 2, 32));
      t = fmaxf(t, __shfl_xor(t, 4, 32));
      t = fmaxf(t, __shfl_xor(t, 8, 32));
      const float mnew = fmaxf(mrun[v], t);

      float cr, e0, e1;
      if (mnew == NINF) { cr = 1.0f; e0 = 0.0f; e1 = 0.0f; }
      else {
        cr = __expf(mrun[v] - mnew);
        e0 = __expf(f0 - mnew);
        e1 = __expf(f1 - mnew);
      }
      float ps = e0 + e1;
      ps += __shfl_xor(ps, 1, 32);
      ps += __shfl_xor(ps, 2, 32);
      ps += __shfl_xor(ps, 4, 32);
      ps += __shfl_xor(ps, 8, 32);
      lrun[v] = lrun[v] * cr + ps;
      mrun[v] = mnew;
      corr[v] = cr;

      Plds[(w * 16 + v + half * 8) * 40 + l16]      = f2bf(e0);
      Plds[(w * 16 + v + half * 8) * 40 + 16 + l16] = f2bf(e1);
    }
#pragma unroll
    for (int jn = 0; jn < 4; ++jn)
#pragma unroll
      for (int v = 0; v < 8; ++v) o[jn][v] *= corr[v];

    __syncthreads();   // make P visible across lanes

    // --- O += P(16x32) @ V(32x64) : 4 WMMAs ---
    AFrag pa;
    {
      const unsigned short* pp = &Plds[(w * 16 + l16) * 40 + half * 8];
      pa.u[0] = *(const uint4*)pp;
      pa.u[1] = *(const uint4*)(pp + 16);
    }
#pragma unroll
    for (int jn = 0; jn < 4; ++jn) {
      AFrag vb;
      const unsigned short* vp = &Vlds[(jn * 16 + l16) * 40 + half * 8];
      vb.u[0] = *(const uint4*)vp;
      vb.u[1] = *(const uint4*)(vp + 16);
      o[jn] = wmma_bf16(pa, vb, o[jn]);
    }
  }

  // --- normalize and store attn output (bf16) ---
#pragma unroll
  for (int v = 0; v < 8; ++v) {
    const int row = qrow0 + v + half * 8;
    const float inv = (lrun[v] > 0.0f) ? 1.0f / lrun[v] : 0.0f;
#pragma unroll
    for (int jn = 0; jn < 4; ++jn)
      Attn[(size_t)(b * Sx + row) * Dx + h * DHx + jn * 16 + l16] =
          f2bf(o[jn][v] * inv);
  }
}

// ---------------------------------------------------------------------------
extern "C" void kernel_launch(void* const* d_in, const int* in_sizes, int n_in,
                              void* d_out, int out_size, void* d_ws, size_t ws_size,
                              hipStream_t stream) {
  (void)in_sizes; (void)n_in; (void)out_size; (void)ws_size;
  const float* x  = (const float*)d_in[0];
  // d_in[1] = mask (recomputed as causal -inf on device)
  const float* Wq = (const float*)d_in[2];
  const float* bq = (const float*)d_in[3];
  const float* Wk = (const float*)d_in[4];
  const float* Wv = (const float*)d_in[5];
  const float* bv = (const float*)d_in[6];
  const float* Wo = (const float*)d_in[7];
  const float* bo = (const float*)d_in[8];

  float* out = (float*)d_out;
  float* qk  = out + (size_t)Bx * Sx * Dx;

  // ws: Q | K | V | attn, each B*S*D bf16 (16 MB) -> 64 MB total
  const size_t E = (size_t)Bx * Sx * Dx;
  unsigned short* qws = (unsigned short*)d_ws;
  unsigned short* kws = qws + E;
  unsigned short* vws = kws + E;
  unsigned short* aws = vws + E;

  const float SCALE = 0.35355339059327373f;   // DH^-0.25, folded into Q and K

  const dim3 blk(256);
  const dim3 gg(Dx / 64, (Bx * Sx) / 128);    // (16, 64)

  gemm_bias_kernel<false, false><<<gg, blk, 0, stream>>>(x, Wq, bq,      qws, SCALE);
  gemm_bias_kernel<false, false><<<gg, blk, 0, stream>>>(x, Wk, nullptr, kws, SCALE);
  gemm_bias_kernel<false, false><<<gg, blk, 0, stream>>>(x, Wv, bv,      vws, 1.0f);

  const size_t qkFloats = (size_t)Bx * Hx * Sx * Sx;        // 134217728
  fill_neginf_kernel<<<dim3((unsigned)(qkFloats / (256 * 8))), blk, 0, stream>>>(
      (float4*)qk);

  flash_attn_kernel<<<dim3(Bx * Hx * (Sx / 128)), blk, 0, stream>>>(
      qws, kws, vws, qk, aws);

  gemm_bias_kernel<true, true><<<gg, blk, 0, stream>>>(aws, Wo, bo, out, 1.0f);
}
